// anisotropy_35991825940947
// MI455X (gfx1250) — matrix operand order repaired and checked by reading, not compiled, not run on hardware
//
#include <hip/hip_runtime.h>
#include <hip/hip_bf16.h>

// ---------------------------------------------------------------------------
// CDNA5 (gfx1250) fused GNN forward.  wave32, WMMA bf16 16x16x32, f32 accum.
// 32-row tiles: each wave owns a 2x2 grid of 16x16 accumulators, so every
// (A-frag, B-frag) pair feeds 2 wmma ops -> half the L2 weight traffic.
// ---------------------------------------------------------------------------

typedef __attribute__((ext_vector_type(16))) __bf16 v16bf;
typedef __attribute__((ext_vector_type(8)))  float  v8f;

struct alignas(16) U4 { unsigned a, b, c, d; };   // POD 16-byte chunk

#define EMB 256
#define TM  32          // rows (edges/nodes) per block

__device__ __forceinline__ unsigned short f2bf(float f) {
    unsigned u = __builtin_bit_cast(unsigned, f);
    unsigned r = (u + 0x7FFFu + ((u >> 16) & 1u)) >> 16;   // RNE
    return (unsigned short)r;
}

// A fragment (16x32 bf16, ISA layout): lanes 0-15 -> K {0..7,16..23},
// lanes 16-31 -> K {8..15,24..31}; row M = lane & 15.  Two 16B LDS loads.
__device__ __forceinline__ v16bf load_a_frag(const unsigned short* A, int stride,
                                             int k0, int lane) {
    int m  = lane & 15;
    int kb = (lane & 16) >> 1;                 // 0 or 8
    const unsigned short* p = A + m * stride + k0 + kb;
    union { U4 q[2]; v16bf v; } f;
    f.q[0] = *(const U4*)(p);
    f.q[1] = *(const U4*)(p + 16);             // K chunk at +16
    return f.v;
}

// B fragment (32x16 bf16): lanes 0-15 hold K {0..7,16..23}, lanes 16-31 hold K {8..15,24..31},
// col N = lane & 15, two K packed per VGPR.  Wt is N-major [Nc][K] so a lane
// reads two runs of 8 contiguous bf16, 16 apart.
__device__ __forceinline__ v16bf load_b_frag(const unsigned short* __restrict__ Wt,
                                             int K, int nBase, int k0, int lane) {
    int n  = nBase + (lane & 15);
    int kb = (lane & 16) >> 1;                 // 0 or 8
    const unsigned short* p = Wt + (size_t)n * K + k0 + kb;
    union { U4 q[2]; v16bf v; } f;
    f.q[0] = *(const U4*)(p);
    f.q[1] = *(const U4*)(p + 16);
    return f.v;
}

__device__ __forceinline__ v8f wmma_bf16(v16bf a, v16bf b, v8f c) {
    return __builtin_amdgcn_wmma_f32_16x16x32_bf16(false, a, false, b,
                                                   (short)0, c, false, false);
}

// ---------------------------------------------------------------------------
// Weight prep: W (K x Nc, f32, row-major)  ->  Wt (Nc x K, bf16)
// ---------------------------------------------------------------------------
__global__ void transpose_bf16_kernel(const float* __restrict__ W,
                                      unsigned short* __restrict__ Wt,
                                      int K, int Nc) {
    int idx = blockIdx.x * 256 + threadIdx.x;
    if (idx < K * Nc) {
        int k = idx / Nc, n = idx % Nc;
        Wt[(size_t)n * K + k] = f2bf(W[idx]);
    }
}

__global__ void zero_kernel(float* __restrict__ p, long n) {
    long i = (long)blockIdx.x * 256 + threadIdx.x;
    if (i < n) p[i] = 0.0f;
}

// x_s = element_emb[x]; x_v = ones; x_e = 0.  grid = N blocks x 256 threads.
__global__ void init_nodes_kernel(const int* __restrict__ x,
                                  const float* __restrict__ element_emb,
                                  float* __restrict__ x_s,
                                  unsigned short* __restrict__ x_s_bf,
                                  float* __restrict__ x_v,
                                  float* __restrict__ x_e, int N) {
    int n = blockIdx.x, t = threadIdx.x;
    int el = x[n];
    float v = element_emb[(size_t)el * EMB + t];
    x_s[(size_t)n * EMB + t]    = v;
    x_s_bf[(size_t)n * EMB + t] = f2bf(v);
    if (t < 9) x_v[(size_t)n * 9 + t] = 1.0f;
    if (t < 3) x_e[(size_t)n * 3 + t] = 0.0f;
}

// x_e[tgt] += e_v   (E*3 threads)
__global__ void edge_xe_kernel(const int* __restrict__ ai,
                               const float* __restrict__ e,
                               float* __restrict__ x_e, int E) {
    int idx = blockIdx.x * 256 + threadIdx.x;
    int ei = idx / 3, c = idx % 3;
    if (ei < E) atomicAdd(&x_e[(size_t)ai[2 * ei + 1] * 3 + c], e[4 * ei + c]);
}

// per-graph counts, u_s accum (graph_emb[x]), u_v accum (x_e)
__global__ void graph_stats_kernel(const int* __restrict__ x,
                                   const int* __restrict__ gi,
                                   const float* __restrict__ graph_emb,
                                   const float* __restrict__ x_e,
                                   float* __restrict__ cnt,
                                   float* __restrict__ u_s,
                                   float* __restrict__ u_v, int N) {
    int n = blockIdx.x * 256 + threadIdx.x;
    if (n < N) {
        int g = gi[n], el = x[n];
        atomicAdd(&cnt[g], 1.0f);
        for (int c = 0; c < 3; ++c) {
            atomicAdd(&u_s[g * 3 + c], graph_emb[(size_t)el * 3 + c]);
            atomicAdd(&u_v[g * 3 + c], x_e[(size_t)n * 3 + c]);
        }
    }
}

// ---------------------------------------------------------------------------
// Edge kernel: 32-edge tile per block (8 wave32), wave = 2x2 C tiles.
//   A = [ x_s_bf[src] | e_s(recomputed RBF@W_e) ]   (32x512 bf16 in LDS)
//   m_s = relu(A @ Wm + bm)     (WMMA, f32 accum, kept in LDS)
//   scatter m_s into s_accum[tgt] (f32 atomics, L2-resident)
//   gates = m_s @ Wg + bg;  m_v = einsum(gates, [x_v[src] | e_v]); scatter.
// ---------------------------------------------------------------------------
__global__ __launch_bounds__(256) void edge_msg_kernel(
    const int* __restrict__ ai, const float* __restrict__ e,
    const unsigned short* __restrict__ x_s_bf,
    const float* __restrict__ x_v,
    const unsigned short* __restrict__ Wm_t,   // 256 x 512 bf16 (N-major)
    const float* __restrict__ bm,
    const float* __restrict__ Wg,              // 256 x 12
    const float* __restrict__ bg,
    const float* __restrict__ W_e,             // 10 x 256
    const float* __restrict__ b_e,
    float* __restrict__ s_accum,               // N x 256
    float* __restrict__ v_accum,               // N x 9
    int E) {
    __shared__ __align__(16) unsigned short A[TM * 512];   // 32 KB
    __shared__ __align__(16) float MS[TM * 256];           // 32 KB
    __shared__ float WgS[256 * 12];                        // 12 KB
    __shared__ float G[TM * 12];
    __shared__ int s_src[TM], s_tgt[TM];
    __shared__ float s_ev[TM][3];

    int tid = threadIdx.x;
    int e0  = blockIdx.x * TM;

    for (int i = tid; i < 256 * 12; i += 256) WgS[i] = Wg[i];
    if (tid < TM) {
        int ei = e0 + tid;
        s_src[tid] = (ei < E) ? ai[2 * ei]     : 0;
        s_tgt[tid] = (ei < E) ? ai[2 * ei + 1] : -1;
    }
    if (tid < TM * 3) {
        int r = tid / 3, c = tid % 3;
        int ei = e0 + r;
        s_ev[r][c] = (ei < E) ? e[4 * ei + c] : 0.0f;
    }
    __syncthreads();

    // Build A tile: 2 passes of (row = tid>>4, 16-col group = tid&15)
#pragma unroll
    for (int rr = 0; rr < 2; ++rr) {
        int r  = rr * 16 + (tid >> 4);
        int cg = tid & 15;
        int ei = e0 + r;
        const unsigned short* xp = x_s_bf + (size_t)s_src[r] * EMB + cg * 16;
        *(U4*)&A[r * 512 + cg * 16]     = *(const U4*)(xp);
        *(U4*)&A[r * 512 + cg * 16 + 8] = *(const U4*)(xp + 8);
        // e_s half: RBF(10) @ W_e + b_e, recomputed (bandwidth-free vs HBM)
        float d = (ei < E) ? e[4 * ei + 3] : 0.0f;
        float rbf[10];
#pragma unroll
        for (int k = 0; k < 10; ++k) {
            float dd = d - (8.0f / 9.0f) * (float)k;   // linspace(0,8,10)
            rbf[k] = __expf(-10.0f * dd * dd);
        }
#pragma unroll
        for (int c = 0; c < 16; ++c) {
            int col = cg * 16 + c;
            float acc = b_e[col];
#pragma unroll
            for (int k = 0; k < 10; ++k) acc += rbf[k] * W_e[k * EMB + col];
            A[r * 512 + 256 + col] = f2bf(acc);
        }
    }
    __syncthreads();

    int lane = tid & 31, wave = tid >> 5;
    int n0 = wave * 32;                 // this wave's 32 output columns
    v8f c00 = {}, c01 = {}, c10 = {}, c11 = {};
#pragma unroll 2
    for (int k0 = 0; k0 < 512; k0 += 32) {
        v16bf a0 = load_a_frag(A,            512, k0, lane);   // rows 0..15
        v16bf a1 = load_a_frag(A + 16 * 512, 512, k0, lane);   // rows 16..31
        v16bf b0 = load_b_frag(Wm_t, 512, n0,      k0, lane);
        v16bf b1 = load_b_frag(Wm_t, 512, n0 + 16, k0, lane);
        c00 = wmma_bf16(a0, b0, c00);
        c01 = wmma_bf16(a0, b1, c01);
        c10 = wmma_bf16(a1, b0, c10);
        c11 = wmma_bf16(a1, b1, c11);
    }
    // epilogue: bias + relu, keep m_s in LDS, scatter to s_accum[tgt]
    int nl = lane & 15;
    int mB = (lane & 16) >> 1;          // lanes 0-15: M=r, 16-31: M=8+r
    int col0 = n0 + nl, col1 = n0 + 16 + nl;
    float bm0 = bm[col0], bm1 = bm[col1];
#pragma unroll
    for (int r = 0; r < 8; ++r) {
        int m0 = mB + r, m1 = 16 + mB + r;
        float v00 = fmaxf(c00[r] + bm0, 0.0f);
        float v01 = fmaxf(c01[r] + bm1, 0.0f);
        float v10 = fmaxf(c10[r] + bm0, 0.0f);
        float v11 = fmaxf(c11[r] + bm1, 0.0f);
        MS[m0 * 256 + col0] = v00;
        MS[m0 * 256 + col1] = v01;
        MS[m1 * 256 + col0] = v10;
        MS[m1 * 256 + col1] = v11;
        int t0 = s_tgt[m0], t1 = s_tgt[m1];
        if (t0 >= 0) {
            atomicAdd(&s_accum[(size_t)t0 * EMB + col0], v00);
            atomicAdd(&s_accum[(size_t)t0 * EMB + col1], v01);
        }
        if (t1 >= 0) {
            atomicAdd(&s_accum[(size_t)t1 * EMB + col0], v10);
            atomicAdd(&s_accum[(size_t)t1 * EMB + col1], v11);
        }
    }
    __syncthreads();

    // gates = m_s @ Wg + bg   (TM edges x 12 outs)
    for (int idx = tid; idx < TM * 12; idx += 256) {
        int r = idx / 12, o = idx % 12;
        float acc = bg[o];
        const float* ms = &MS[r * 256];
        for (int k = 0; k < 256; ++k) acc += ms[k] * WgS[k * 12 + o];
        G[idx] = acc;
    }
    __syncthreads();

    // m_v[e][d][o] = sum_c gates[e][o][c] * basis[e][d][c];
    // basis[:, :, 0:3] = x_v[src], basis[:, :, 3] = e_v
    for (int idx = tid; idx < TM * 9; idx += 256) {
        int r = idx / 9, rem = idx % 9;
        int dd = rem / 3, o = rem % 3;
        int t = s_tgt[r];
        if (t >= 0) {
            const float* g  = &G[r * 12 + o * 4];
            const float* xv = x_v + (size_t)s_src[r] * 9 + dd * 3;
            float mv = g[0] * xv[0] + g[1] * xv[1] + g[2] * xv[2]
                     + g[3] * s_ev[r][dd];
            atomicAdd(&v_accum[(size_t)t * 9 + dd * 3 + o], mv);
        }
    }
}

// ---------------------------------------------------------------------------
// Node kernel: new_s = relu(s_accum @ Wu + bu) [+ x_s]; new_v = v_accum [+ x_v]
// 32-node tile per block, WMMA over K=256, 2x2 tiles per wave.
// ---------------------------------------------------------------------------
__global__ __launch_bounds__(256) void node_update_kernel(
    const float* __restrict__ s_accum,
    const unsigned short* __restrict__ Wu_t,   // 256 x 256 bf16 (N-major)
    const float* __restrict__ bu,
    const float* __restrict__ v_accum,
    float* __restrict__ x_s,
    unsigned short* __restrict__ x_s_bf,
    float* __restrict__ x_v,
    int N, int residual) {
    __shared__ __align__(16) unsigned short A[TM * 256];   // 16 KB
    int tid = threadIdx.x;
    int n0  = blockIdx.x * TM;
#pragma unroll
    for (int rr = 0; rr < 2; ++rr) {
        int r = rr * 16 + (tid >> 4), cg = tid & 15;
        int ni = n0 + r;
        const float* sp = s_accum + (size_t)(ni < N ? ni : 0) * EMB + cg * 16;
#pragma unroll
        for (int c = 0; c < 16; ++c) A[r * 256 + cg * 16 + c] = f2bf(sp[c]);
    }
    __syncthreads();

    int lane = tid & 31, wave = tid >> 5;
    int nb = wave * 32;
    v8f c00 = {}, c01 = {}, c10 = {}, c11 = {};
#pragma unroll 2
    for (int k0 = 0; k0 < 256; k0 += 32) {
        v16bf a0 = load_a_frag(A,            256, k0, lane);
        v16bf a1 = load_a_frag(A + 16 * 256, 256, k0, lane);
        v16bf b0 = load_b_frag(Wu_t, 256, nb,      k0, lane);
        v16bf b1 = load_b_frag(Wu_t, 256, nb + 16, k0, lane);
        c00 = wmma_bf16(a0, b0, c00);
        c01 = wmma_bf16(a0, b1, c01);
        c10 = wmma_bf16(a1, b0, c10);
        c11 = wmma_bf16(a1, b1, c11);
    }
    int nl = lane & 15;
    int mB = (lane & 16) >> 1;
    int col0 = nb + nl, col1 = nb + 16 + nl;
    float bu0 = bu[col0], bu1 = bu[col1];
#pragma unroll
    for (int r = 0; r < 8; ++r) {
#pragma unroll
        for (int h = 0; h < 2; ++h) {
            int ni = n0 + h * 16 + mB + r;
            if (ni < N) {
                float v0 = fmaxf((h ? c10[r] : c00[r]) + bu0, 0.0f);
                float v1 = fmaxf((h ? c11[r] : c01[r]) + bu1, 0.0f);
                if (residual) {
                    v0 += x_s[(size_t)ni * EMB + col0];
                    v1 += x_s[(size_t)ni * EMB + col1];
                }
                x_s[(size_t)ni * EMB + col0] = v0;
                x_s[(size_t)ni * EMB + col1] = v1;
                x_s_bf[(size_t)ni * EMB + col0] = f2bf(v0);
                x_s_bf[(size_t)ni * EMB + col1] = f2bf(v1);
            }
        }
    }
    for (int idx = tid; idx < TM * 9; idx += 256) {
        int r = idx / 9, c = idx % 9;
        int ni = n0 + r;
        if (ni < N) {
            float v = v_accum[(size_t)ni * 9 + c];
            if (residual) v += x_v[(size_t)ni * 9 + c];
            x_v[(size_t)ni * 9 + c] = v;
        }
    }
}

// pooled sums over graphs (grid = N blocks x 256)
__global__ void pool_kernel(const int* __restrict__ gi,
                            const float* __restrict__ x_s,
                            const float* __restrict__ x_v,
                            float* __restrict__ pooled_s,
                            float* __restrict__ pooled_v, int N) {
    int n = blockIdx.x, t = threadIdx.x;
    int g = gi[n];
    atomicAdd(&pooled_s[(size_t)g * EMB + t], x_s[(size_t)n * EMB + t]);
    if (t < 9) atomicAdd(&pooled_v[(size_t)g * 9 + t], x_v[(size_t)n * 9 + t]);
}

// final readout: one thread per graph
__global__ void out_kernel(const float* __restrict__ pooled_s,
                           const float* __restrict__ pooled_v,
                           const float* __restrict__ u_s,
                           const float* __restrict__ u_v,
                           const float* __restrict__ cnt,
                           const float* __restrict__ W_v,   // 4 x 3
                           const float* __restrict__ W_s,   // 259 x 3
                           const float* __restrict__ b_s,
                           float* __restrict__ out, int B) {
    int b = threadIdx.x;
    if (b < B) {
        float ic = 1.0f / fmaxf(cnt[b], 1.0f);
        float os[3];
        for (int t = 0; t < 3; ++t) {
            float acc = b_s[t];
            for (int k = 0; k < EMB; ++k)
                acc += pooled_s[(size_t)b * EMB + k] * ic * W_s[k * 3 + t];
            for (int c = 0; c < 3; ++c)
                acc += u_s[b * 3 + c] * ic * W_s[(EMB + c) * 3 + t];
            os[t] = acc;
        }
        for (int d = 0; d < 3; ++d) {
            for (int o = 0; o < 3; ++o) {
                float acc = 0.0f;
                for (int c = 0; c < 3; ++c)
                    acc += pooled_v[(size_t)b * 9 + d * 3 + c] * ic * W_v[c * 3 + o];
                acc += u_v[b * 3 + d] * ic * W_v[3 * 3 + o];
                out[b * 12 + d * 4 + o] = acc;
            }
            out[b * 12 + d * 4 + 3] = os[d];
        }
    }
}

// ---------------------------------------------------------------------------
extern "C" void kernel_launch(void* const* d_in, const int* in_sizes, int n_in,
                              void* d_out, int out_size, void* d_ws, size_t ws_size,
                              hipStream_t stream) {
    const int*   x   = (const int*)d_in[0];
    const int*   ai  = (const int*)d_in[1];
    const float* e   = (const float*)d_in[2];
    const int*   gi  = (const int*)d_in[3];
    const float* element_emb = (const float*)d_in[5];
    const float* graph_emb   = (const float*)d_in[6];
    const float* W_e = (const float*)d_in[7];
    const float* b_e = (const float*)d_in[8];
    const float *Wm[4], *bm[4], *Wg[4], *bg[4], *Wu[4], *bu[4];
    for (int j = 0; j < 4; ++j) {
        Wm[j] = (const float*)d_in[9 + 6 * j + 0];
        bm[j] = (const float*)d_in[9 + 6 * j + 1];
        Wg[j] = (const float*)d_in[9 + 6 * j + 2];
        bg[j] = (const float*)d_in[9 + 6 * j + 3];
        Wu[j] = (const float*)d_in[9 + 6 * j + 4];
        bu[j] = (const float*)d_in[9 + 6 * j + 5];
    }
    const float* W_v = (const float*)d_in[33];
    const float* W_s = (const float*)d_in[34];
    const float* b_s = (const float*)d_in[35];
    float* out = (float*)d_out;

    const int N = in_sizes[0];
    const int E = in_sizes[1] / 2;
    const int B = 128;

    // workspace carving (256B aligned)
    char* ws = (char*)d_ws;
    size_t off = 0;
    auto carve = [&](size_t bytes) -> void* {
        off = (off + 255) & ~(size_t)255;
        void* p = ws + off;
        off += bytes;
        return p;
    };
    unsigned short* Wm_t    = (unsigned short*)carve((size_t)4 * 256 * 512 * 2);
    unsigned short* Wu_t    = (unsigned short*)carve((size_t)4 * 256 * 256 * 2);
    float*          x_s     = (float*)carve((size_t)N * EMB * 4);
    unsigned short* x_s_bf  = (unsigned short*)carve((size_t)N * EMB * 2);
    float*          s_accum = (float*)carve((size_t)N * EMB * 4);
    float*          x_v     = (float*)carve((size_t)N * 9 * 4);
    float*          v_accum = (float*)carve((size_t)N * 9 * 4);
    float*          x_e     = (float*)carve((size_t)N * 3 * 4);
    float*          pooled_s = (float*)carve((size_t)B * EMB * 4);
    float*          pooled_v = (float*)carve((size_t)B * 9 * 4);
    float*          u_s      = (float*)carve((size_t)B * 3 * 4);
    float*          u_v      = (float*)carve((size_t)B * 3 * 4);
    float*          cnt      = (float*)carve((size_t)B * 4);

    // 1) weight prep (bf16, N-major) — L2 resident thereafter
    for (int j = 0; j < 4; ++j) {
        transpose_bf16_kernel<<<(512 * 256 + 255) / 256, 256, 0, stream>>>(
            Wm[j], Wm_t + (size_t)j * 256 * 512, 512, 256);
        transpose_bf16_kernel<<<(256 * 256 + 255) / 256, 256, 0, stream>>>(
            Wu[j], Wu_t + (size_t)j * 256 * 256, 256, 256);
    }
    // 2) node init + graph stats
    init_nodes_kernel<<<N, 256, 0, stream>>>(x, element_emb, x_s, x_s_bf, x_v, x_e, N);
    zero_kernel<<<((long)B * EMB + 255) / 256, 256, 0, stream>>>(pooled_s, (long)B * EMB);
    zero_kernel<<<((long)B * 9 + 255) / 256, 256, 0, stream>>>(pooled_v, (long)B * 9);
    zero_kernel<<<1, 256, 0, stream>>>(u_s, (long)B * 3);
    zero_kernel<<<1, 256, 0, stream>>>(u_v, (long)B * 3);
    zero_kernel<<<1, 256, 0, stream>>>(cnt, (long)B);
    edge_xe_kernel<<<((long)E * 3 + 255) / 256, 256, 0, stream>>>(ai, e, x_e, E);
    graph_stats_kernel<<<(N + 255) / 256, 256, 0, stream>>>(x, gi, graph_emb, x_e,
                                                            cnt, u_s, u_v, N);
    // 3) 4 message-passing layers
    for (int j = 0; j < 4; ++j) {
        zero_kernel<<<((long)N * EMB + 255) / 256, 256, 0, stream>>>(s_accum, (long)N * EMB);
        zero_kernel<<<((long)N * 9 + 255) / 256, 256, 0, stream>>>(v_accum, (long)N * 9);
        edge_msg_kernel<<<(E + TM - 1) / TM, 256, 0, stream>>>(
            ai, e, x_s_bf, x_v, Wm_t + (size_t)j * 256 * 512, bm[j], Wg[j], bg[j],
            W_e, b_e, s_accum, v_accum, E);
        node_update_kernel<<<(N + TM - 1) / TM, 256, 0, stream>>>(
            s_accum, Wu_t + (size_t)j * 256 * 256, bu[j], v_accum,
            x_s, x_s_bf, x_v, N, j > 0 ? 1 : 0);
    }
    // 4) pooling + readout
    pool_kernel<<<N, 256, 0, stream>>>(gi, x_s, x_v, pooled_s, pooled_v, N);
    out_kernel<<<1, 128, 0, stream>>>(pooled_s, pooled_v, u_s, u_v, cnt,
                                      W_v, W_s, b_s, out, B);
    (void)n_in; (void)out_size; (void)ws_size;
}
